// SparseMoEBlock_25872882991286
// MI455X (gfx1250) — compile-verified, matrix operand order checked by reading, not run on hardware
//
#include <hip/hip_runtime.h>

// ---------------------------------------------------------------------------
// SparseMoEBlock for MI455X (gfx1250): bf16 WMMA GEMMs + fp32 accumulate,
// async GLOBAL->LDS staging (ASYNCcnt) for the big SwiGLU GEMMs.
// B=2, S=1024, H=1024, I=512, SI=2048, E=8, TOPK=2
// ---------------------------------------------------------------------------

typedef __bf16 bf16;
typedef __attribute__((ext_vector_type(16))) __bf16 v16bf;
typedef __attribute__((ext_vector_type(8)))  __bf16 v8bf;
typedef __attribute__((ext_vector_type(8)))  float  v8f;
typedef int v4i __attribute__((vector_size(16)));   // matches async-LDS builtin

constexpr int T_TOK = 2048;   // B*S
constexpr int HDIM  = 1024;
constexpr int IDIM  = 512;
constexpr int SIDIM = 2048;
constexpr int NEXP  = 8;

constexpr int KCH   = 32;     // K-chunk depth per WMMA stage
constexpr int APAD  = 40;     // LDS row stride (32 data + 8 pad) -> conflict-free

#if __has_builtin(__builtin_amdgcn_global_load_async_to_lds_b128) && \
    __has_builtin(__builtin_amdgcn_s_wait_asynccnt)
#define HAVE_ASYNC_LDS 1
#else
#define HAVE_ASYNC_LDS 0
#endif

#define AS_GLOBAL __attribute__((address_space(1)))
#define AS_LDS    __attribute__((address_space(3)))

__device__ __forceinline__ v8f wmma_bf16(v16bf a, v16bf b, v8f c) {
    // D = A(16x32 bf16) * B(32x16 bf16) + C(16x16 f32)
    return __builtin_amdgcn_wmma_f32_16x16x32_bf16(
        /*neg_a=*/false, a, /*neg_b=*/false, b,
        /*c_mod=*/(short)0, c, /*reuse_a=*/false, /*reuse_b=*/false);
}

__device__ __forceinline__ float silu_f(float x) {
    return x / (1.0f + __expf(-x));
}

// 16-byte GLOBAL -> LDS copy: async (no VGPR round-trip) when available.
__device__ __forceinline__ void copy16_g2l(const bf16* g, bf16* l) {
#if HAVE_ASYNC_LDS
    __builtin_amdgcn_global_load_async_to_lds_b128(
        (AS_GLOBAL v4i*)g, (AS_LDS v4i*)l, /*offset=*/0, /*cpol=*/0);
#else
    *(v8bf*)l = *(const v8bf*)g;
#endif
}

__device__ __forceinline__ void wait_async_4() {
#if HAVE_ASYNC_LDS
    __builtin_amdgcn_s_wait_asynccnt(4);
#endif
}
__device__ __forceinline__ void wait_async_0() {
#if HAVE_ASYNC_LDS
    __builtin_amdgcn_s_wait_asynccnt(0);
#endif
}

// Pin the hot-loop schedule: a burst of DS reads, then a burst of WMMAs.
__device__ __forceinline__ void sched_ds_then_wmma() {
#if __has_builtin(__builtin_amdgcn_sched_group_barrier)
    __builtin_amdgcn_sched_group_barrier(0x100, 18, 0);  // 18x DS read
    __builtin_amdgcn_sched_group_barrier(0x008,  8, 0);  //  8x WMMA (MFMA class)
#elif __has_builtin(__builtin_amdgcn_sched_barrier)
    __builtin_amdgcn_sched_barrier(0);
#endif
}

// ---------------------------------------------------------------------------
// fp32 -> bf16 conversion (grid-stride)
// ---------------------------------------------------------------------------
__global__ void f32_to_bf16_kernel(const float* __restrict__ src,
                                   bf16* __restrict__ dst, long n) {
    long i = (long)blockIdx.x * blockDim.x + threadIdx.x;
    long stride = (long)gridDim.x * blockDim.x;
    for (; i < n; i += stride) dst[i] = (bf16)src[i];
}

// ---------------------------------------------------------------------------
// Router: softmax over E=8 logits, top-2, renormalize -> cw[T,8];
// also the shared-expert sigmoid gate g[T]. One wave32 per token.
// ---------------------------------------------------------------------------
__global__ void router_kernel(const float* __restrict__ x,     // [T,H]
                              const float* __restrict__ rw,    // [E,H]
                              const float* __restrict__ gw,    // [1,H]
                              float* __restrict__ cw,          // [T,E]
                              float* __restrict__ gout) {      // [T]
    int lane = threadIdx.x & 31;
    int tok  = blockIdx.x * (blockDim.x >> 5) + (threadIdx.x >> 5);
    if (tok >= T_TOK) return;

    float acc[NEXP];
#pragma unroll
    for (int e = 0; e < NEXP; ++e) acc[e] = 0.0f;
    float gacc = 0.0f;

    const float* xr = x + (size_t)tok * HDIM;
    for (int h = lane; h < HDIM; h += 32) {
        float xv = xr[h];
#pragma unroll
        for (int e = 0; e < NEXP; ++e) acc[e] += xv * rw[e * HDIM + h];
        gacc += xv * gw[h];
    }
#pragma unroll
    for (int e = 0; e < NEXP; ++e)
        for (int off = 16; off > 0; off >>= 1)
            acc[e] += __shfl_xor(acc[e], off, 32);
    for (int off = 16; off > 0; off >>= 1)
        gacc += __shfl_xor(gacc, off, 32);

    if (lane == 0) {
        float m = acc[0];
#pragma unroll
        for (int e = 1; e < NEXP; ++e) m = fmaxf(m, acc[e]);
        float p[NEXP];
        float sum = 0.0f;
#pragma unroll
        for (int e = 0; e < NEXP; ++e) { p[e] = __expf(acc[e] - m); sum += p[e]; }
        float inv_sum = 1.0f / sum;
#pragma unroll
        for (int e = 0; e < NEXP; ++e) p[e] *= inv_sum;

        int   i1 = 0; float v1 = p[0];
#pragma unroll
        for (int e = 1; e < NEXP; ++e) if (p[e] > v1) { v1 = p[e]; i1 = e; }
        int   i2 = -1; float v2 = -1.0f;
#pragma unroll
        for (int e = 0; e < NEXP; ++e)
            if (e != i1 && p[e] > v2) { v2 = p[e]; i2 = e; }

        float inv = 1.0f / (v1 + v2);
#pragma unroll
        for (int e = 0; e < NEXP; ++e)
            cw[(size_t)tok * NEXP + e] =
                (e == i1) ? v1 * inv : ((e == i2) ? v2 * inv : 0.0f);
        gout[tok] = 1.0f / (1.0f + __expf(-gacc));
    }
}

// ---------------------------------------------------------------------------
// SwiGLU up-projection GEMM with async double-buffered LDS staging.
//   act[t,n] = silu(x @ Wg[n,:]^T) * (x @ Wu[n,:]^T)
// Workgroup (256 thr = 8 waves) owns a 128(M) x 64(N) macro-tile; wave w
// computes rows [w*16, w*16+16). Per 32-deep K chunk, each thread issues
// exactly 4 global_load_async_to_lds_b128 (2 A + 1 gate + 1 up); pipeline
// waits ASYNCcnt<=4 so the chunk being consumed is complete while the next
// chunk's loads fly behind the WMMAs. sched_group_barrier pins the inner
// schedule to [18x ds_load_b128][8x WMMA] so there is a single dscnt wait
// per chunk instead of one per WMMA. LDS rows padded to 40 elements (80B
// stride) so the half-wave ds_load_b128 pattern tiles all 64 banks once.
// blockIdx.y = expert (stride offsets); shared expert uses gridDim.y=1.
// ---------------------------------------------------------------------------
__global__ __launch_bounds__(256)
void act_gemm_lds_kernel(const bf16* __restrict__ x,   // [T,K]
                         const bf16* __restrict__ Wg,  // rows n -> +n*K
                         const bf16* __restrict__ Wu,
                         bf16* __restrict__ outp,      // [T,N] bf16
                         int N, int K,
                         long wg_es, long wu_es, long out_es) {
    __shared__ bf16 As[2][128 * APAD];
    __shared__ bf16 Gs[2][64 * APAD];
    __shared__ bf16 Us[2][64 * APAD];

    long e = blockIdx.y;
    Wg   += e * wg_es;
    Wu   += e * wu_es;
    outp += e * out_es;

    int nstrips = N >> 6;
    int bm  = blockIdx.x / nstrips;
    int bn  = blockIdx.x % nstrips;
    int tm0 = bm * 128;
    int n0  = bn * 64;

    int tid  = threadIdx.x;
    int wid  = tid >> 5;
    int lane = tid & 31;
    int lh   = lane & 15;          // row/col within 16-tile
    int kk   = (lane >> 4) << 4;   // 0 or 16: K sub-offset for this half-wave

    // staging assignment: 16B chunk id -> (row, chunk-pos)
    int srow = tid >> 2;            // 0..63
    int scp  = (tid & 3) * 8;       // element offset within 32-deep chunk

    const bf16* Ag0 = x  + (size_t)(tm0 + srow) * K + scp;
    const bf16* Ag1 = x  + (size_t)(tm0 + 64 + srow) * K + scp;
    const bf16* Gg  = Wg + (size_t)(n0 + srow) * K + scp;
    const bf16* Ug  = Wu + (size_t)(n0 + srow) * K + scp;

    auto issue = [&](int kc, int b) {
        int k0 = kc * KCH;
        copy16_g2l(Ag0 + k0, &As[b][srow * APAD + scp]);
        copy16_g2l(Ag1 + k0, &As[b][(64 + srow) * APAD + scp]);
        copy16_g2l(Gg  + k0, &Gs[b][srow * APAD + scp]);
        copy16_g2l(Ug  + k0, &Us[b][srow * APAD + scp]);
    };

    v8f accg[4] = {};
    v8f accu[4] = {};

    int nk = K / KCH;
    issue(0, 0);
    for (int kc = 0; kc < nk; ++kc) {
        int b = kc & 1;
        if (kc + 1 < nk) {
            issue(kc + 1, (kc + 1) & 1);
            wait_async_4();          // chunk kc complete; kc+1 still in flight
        } else {
            wait_async_0();
        }
        __syncthreads();

        // All tile reads first, then the WMMA burst (pinned via
        // sched_group_barrier below).
        v16bf a = *(const v16bf*)&As[b][(wid * 16 + lh) * APAD + kk];
        v16bf bg[4], bu[4];
#pragma unroll
        for (int j = 0; j < 4; ++j) {
            bg[j] = *(const v16bf*)&Gs[b][(j * 16 + lh) * APAD + kk];
            bu[j] = *(const v16bf*)&Us[b][(j * 16 + lh) * APAD + kk];
        }
#pragma unroll
        for (int j = 0; j < 4; ++j) {
            accg[j] = wmma_bf16(a, bg[j], accg[j]);
            accu[j] = wmma_bf16(a, bu[j], accu[j]);
        }
        sched_ds_then_wmma();        // [18x DS read][8x WMMA]
        __syncthreads();             // protect buffer b from chunk kc+2 writes
    }

    // C/D layout: VGPR r, lanes 0-15 -> M=r ; lanes 16-31 -> M=8+r ; N=lane&15
    int rbase = (lane >> 4) << 3;
    int col   = lane & 15;
    int t0    = tm0 + wid * 16;
#pragma unroll
    for (int j = 0; j < 4; ++j)
#pragma unroll
        for (int r = 0; r < 8; ++r) {
            float gv = accg[j][r];
            float uv = accu[j][r];
            outp[(size_t)(t0 + rbase + r) * N + n0 + j * 16 + col] =
                (bf16)(silu_f(gv) * uv);
        }
}

// ---------------------------------------------------------------------------
// MoE down-projection + masked combine:
//   out[t,h] = sum_e cw[t,e] * sum_i act[e,t,i] * Wd[e,h,i]
// Per-expert f32 partial accumulator, row-scaled by cw before merging.
// ---------------------------------------------------------------------------
__global__ void moe_down_kernel(const bf16* __restrict__ act,  // [E,T,I]
                                const bf16* __restrict__ Wd,   // [E,H,I]
                                const float* __restrict__ cw,  // [T,E]
                                float* __restrict__ outp) {    // [T,H]
    int nstrips = HDIM >> 6;   // 16
    int wave  = blockIdx.x * (blockDim.x >> 5) + (threadIdx.x >> 5);
    int total = (T_TOK / 16) * nstrips;
    if (wave >= total) return;
    int m  = wave / nstrips, ns = wave % nstrips;
    int t0 = m * 16,         n0 = ns * 64;

    int lane  = threadIdx.x & 31;
    int lh    = lane & 15;
    int kk    = (lane >> 4) << 4;
    int rbase = (lane >> 4) << 3;
    int col   = lane & 15;

    v8f acc[4] = {};
    for (int e = 0; e < NEXP; ++e) {
        v8f part[4] = {};
        const bf16* Ab = act + ((size_t)e * T_TOK + t0 + lh) * IDIM + kk;
        const bf16* Bb = Wd  + ((size_t)e * HDIM  + n0 + lh) * IDIM + kk;
        for (int k0 = 0; k0 < IDIM; k0 += 32) {
            v16bf a = *(const v16bf*)(Ab + k0);
#pragma unroll
            for (int j = 0; j < 4; ++j) {
                v16bf b = *(const v16bf*)(Bb + (size_t)j * 16 * IDIM + k0);
                part[j] = wmma_bf16(a, b, part[j]);
            }
        }
#pragma unroll
        for (int r = 0; r < 8; ++r) {
            float cwv = cw[(size_t)(t0 + rbase + r) * NEXP + e];
#pragma unroll
            for (int j = 0; j < 4; ++j) acc[j][r] += cwv * part[j][r];
        }
    }

#pragma unroll
    for (int j = 0; j < 4; ++j)
#pragma unroll
        for (int r = 0; r < 8; ++r)
            outp[(size_t)(t0 + rbase + r) * HDIM + n0 + j * 16 + col] = acc[j][r];
}

// ---------------------------------------------------------------------------
// Shared-expert down-projection + gated add:
//   out[t,h] += g[t] * sum_s act_sh[t,s] * Wd_sh[h,s]
// Must run after moe_down_kernel (stream-ordered RMW on d_out).
// ---------------------------------------------------------------------------
__global__ void shared_down_kernel(const bf16* __restrict__ act,  // [T,SI]
                                   const bf16* __restrict__ Wd,   // [H,SI]
                                   const float* __restrict__ g,   // [T]
                                   float* __restrict__ outp) {    // [T,H]
    int nstrips = HDIM >> 6;
    int wave  = blockIdx.x * (blockDim.x >> 5) + (threadIdx.x >> 5);
    int total = (T_TOK / 16) * nstrips;
    if (wave >= total) return;
    int m  = wave / nstrips, ns = wave % nstrips;
    int t0 = m * 16,         n0 = ns * 64;

    int lane  = threadIdx.x & 31;
    int lh    = lane & 15;
    int kk    = (lane >> 4) << 4;
    int rbase = (lane >> 4) << 3;
    int col   = lane & 15;

    const bf16* Ab = act + (size_t)(t0 + lh) * SIDIM + kk;
    const bf16* Bb = Wd  + (size_t)(n0 + lh) * SIDIM + kk;

    v8f acc[4] = {};
    for (int k0 = 0; k0 < SIDIM; k0 += 32) {
        v16bf a = *(const v16bf*)(Ab + k0);
#pragma unroll
        for (int j = 0; j < 4; ++j) {
            v16bf b = *(const v16bf*)(Bb + (size_t)j * 16 * SIDIM + k0);
            acc[j] = wmma_bf16(a, b, acc[j]);
        }
    }

#pragma unroll
    for (int r = 0; r < 8; ++r) {
        float gt = g[t0 + rbase + r];
#pragma unroll
        for (int j = 0; j < 4; ++j) {
            size_t idx = (size_t)(t0 + rbase + r) * HDIM + n0 + j * 16 + col;
            outp[idx] += gt * acc[j][r];
        }
    }
}

// ---------------------------------------------------------------------------
// Host-side launcher
// ---------------------------------------------------------------------------
extern "C" void kernel_launch(void* const* d_in, const int* in_sizes, int n_in,
                              void* d_out, int out_size, void* d_ws, size_t ws_size,
                              hipStream_t stream) {
    const float* x_f   = (const float*)d_in[0];  // [T,H]
    const float* rw_f  = (const float*)d_in[1];  // [E,H]
    const float* gu_f  = (const float*)d_in[2];  // [E,2I,H]
    const float* dn_f  = (const float*)d_in[3];  // [E,H,I]
    const float* sg_f  = (const float*)d_in[4];  // [SI,H]
    const float* su_f  = (const float*)d_in[5];  // [SI,H]
    const float* sd_f  = (const float*)d_in[6];  // [H,SI]
    const float* seg_f = (const float*)d_in[7];  // [1,H]
    float* out = (float*)d_out;

    // ---- workspace carve-up (bf16 copies + activations + router outputs) ----
    char*  ws  = (char*)d_ws;
    size_t off = 0;
    auto alloc = [&](size_t bytes) -> void* {
        void* p = ws + off;
        off = (off + bytes + 255) & ~(size_t)255;
        return p;
    };
    const long n_x  = (long)T_TOK * HDIM;
    const long n_gu = (long)NEXP * 2 * IDIM * HDIM;
    const long n_dn = (long)NEXP * HDIM * IDIM;
    const long n_sg = (long)SIDIM * HDIM;
    const long n_sd = (long)HDIM * SIDIM;

    bf16* x_bf   = (bf16*)alloc(n_x  * sizeof(bf16));
    bf16* gu_bf  = (bf16*)alloc(n_gu * sizeof(bf16));
    bf16* dn_bf  = (bf16*)alloc(n_dn * sizeof(bf16));
    bf16* sg_bf  = (bf16*)alloc(n_sg * sizeof(bf16));
    bf16* su_bf  = (bf16*)alloc(n_sg * sizeof(bf16));
    bf16* sd_bf  = (bf16*)alloc(n_sd * sizeof(bf16));
    bf16* act_sh = (bf16*)alloc((size_t)T_TOK * SIDIM * sizeof(bf16));
    bf16* act_mo = (bf16*)alloc((size_t)NEXP * T_TOK * IDIM * sizeof(bf16));
    float* cw    = (float*)alloc((size_t)T_TOK * NEXP * sizeof(float));
    float* gvec  = (float*)alloc((size_t)T_TOK * sizeof(float));
    (void)ws_size; (void)in_sizes; (void)n_in; (void)out_size;

    // 1) fp32 -> bf16 conversions
    f32_to_bf16_kernel<<<2048, 256, 0, stream>>>(x_f,  x_bf,  n_x);
    f32_to_bf16_kernel<<<2048, 256, 0, stream>>>(gu_f, gu_bf, n_gu);
    f32_to_bf16_kernel<<<2048, 256, 0, stream>>>(dn_f, dn_bf, n_dn);
    f32_to_bf16_kernel<<<2048, 256, 0, stream>>>(sg_f, sg_bf, n_sg);
    f32_to_bf16_kernel<<<2048, 256, 0, stream>>>(su_f, su_bf, n_sg);
    f32_to_bf16_kernel<<<2048, 256, 0, stream>>>(sd_f, sd_bf, n_sd);

    // 2) router (fp32 precision) -> cw[T,8], gvec[T]
    router_kernel<<<T_TOK / 4, 128, 0, stream>>>(x_f, rw_f, seg_f, cw, gvec);

    // 3) shared-expert SwiGLU: act_sh[T,SI]  (async-LDS staged)
    {
        dim3 grid((T_TOK / 128) * (SIDIM / 64), 1);   // 16*32 = 512
        act_gemm_lds_kernel<<<grid, 256, 0, stream>>>(
            x_bf, sg_bf, su_bf, act_sh, SIDIM, HDIM, 0, 0, 0);
    }

    // 4) per-expert SwiGLU: act_mo[E,T,I]  (async-LDS staged)
    {
        dim3 grid((T_TOK / 128) * (IDIM / 64), NEXP); // 16*8 = 128 x 8
        act_gemm_lds_kernel<<<grid, 256, 0, stream>>>(
            x_bf, gu_bf, gu_bf + (size_t)IDIM * HDIM, act_mo, IDIM, HDIM,
            (long)2 * IDIM * HDIM, (long)2 * IDIM * HDIM, (long)T_TOK * IDIM);
    }

    // 5) MoE down + combine -> d_out (full overwrite)
    {
        int waves = (T_TOK / 16) * (HDIM / 64);
        moe_down_kernel<<<waves / 8, 256, 0, stream>>>(act_mo, dn_bf, cw, out);
    }

    // 6) shared down + sigmoid gate, accumulate into d_out
    {
        int waves = (T_TOK / 16) * (HDIM / 64);
        shared_down_kernel<<<waves / 8, 256, 0, stream>>>(act_sh, sd_bf, gvec, out);
    }
}
